// VanillaRNN_50663434223724
// MI455X (gfx1250) — compile-verified, matrix operand order checked by reading, not run on hardware
//
#include <hip/hip_runtime.h>
#include <hip/hip_fp16.h>

typedef __attribute__((ext_vector_type(16))) _Float16 v16h;
typedef __attribute__((ext_vector_type(8)))  _Float16 v8h;
typedef __attribute__((ext_vector_type(8)))  float    v8f;

#define SEQ 8192
#define HID 2048
#define DIN 2048

// ---------------------------------------------------------------------------
// Phase 1: P = X (SEQ x DIN) * Wxh^T (HID x DIN, row-major) + b   -> d_out
// f16 WMMA, f32 accumulate. Block tile 128(M) x 64(N), 8 waves of 32.
// ---------------------------------------------------------------------------
#define BM 128
#define BN 64
#define BK 32

// Load a 16x32 f16 fragment from LDS (row-major, stride 32 halves) in the
// CDNA5 16-bit A layout: lane<16 -> K 0..7 & 16..23 ; lane>=16 -> K 8..15 & 24..31.
__device__ inline v16h load_frag16(const _Float16* base, int row, int half) {
  union { v16h v; v8h h[2]; } u;
  const _Float16* p = base + row * BK;
  u.h[0] = *(const v8h*)(p + half * 8);        // ds_load_b128
  u.h[1] = *(const v8h*)(p + 16 + half * 8);   // ds_load_b128
  return u.v;
}

__global__ __launch_bounds__(256)
void gemm_xh_kernel(const float* __restrict__ X, const float* __restrict__ W,
                    const float* __restrict__ bias, float* __restrict__ P)
{
  __shared__ __align__(16) _Float16 As[BM][BK];  // 8 KB
  __shared__ __align__(16) _Float16 Bs[BN][BK];  // 4 KB

  const int tid  = threadIdx.x;
  const int wid  = tid >> 5;
  const int lane = tid & 31;
  const int half = lane >> 4;
  const int l16  = lane & 15;

  const int tileM = blockIdx.x * BM;
  const int tileN = blockIdx.y * BN;
  const int wm = (wid & 3) * 32;   // wave M offset (4 waves over 128 rows)
  const int wn = (wid >> 2) * 32;  // wave N offset (2 waves over 64 cols)

  v8f acc[2][2];
#pragma unroll
  for (int i = 0; i < 2; ++i)
#pragma unroll
    for (int j = 0; j < 2; ++j)
#pragma unroll
      for (int r = 0; r < 8; ++r) acc[i][j][r] = 0.0f;

  for (int kk = 0; kk < DIN; kk += BK) {
    // Stage A tile: 128x32 f32 -> f16.  1024 float4 chunks, 4 per thread.
#pragma unroll
    for (int i = 0; i < 4; ++i) {
      int c  = tid + 256 * i;          // 0..1023
      int r  = c >> 3;                 // 0..127
      int k4 = (c & 7) * 4;            // 0..28
      float4 f = *(const float4*)(X + (size_t)(tileM + r) * DIN + kk + k4);
      As[r][k4 + 0] = (_Float16)f.x;  As[r][k4 + 1] = (_Float16)f.y;
      As[r][k4 + 2] = (_Float16)f.z;  As[r][k4 + 3] = (_Float16)f.w;
    }
    // Stage B tile: 64x32 f32 -> f16 (W is N x K row-major). 2 chunks/thread.
#pragma unroll
    for (int i = 0; i < 2; ++i) {
      int c  = tid + 256 * i;          // 0..511
      int r  = c >> 3;                 // 0..63
      int k4 = (c & 7) * 4;
      float4 f = *(const float4*)(W + (size_t)(tileN + r) * DIN + kk + k4);
      Bs[r][k4 + 0] = (_Float16)f.x;  Bs[r][k4 + 1] = (_Float16)f.y;
      Bs[r][k4 + 2] = (_Float16)f.z;  Bs[r][k4 + 3] = (_Float16)f.w;
    }
    __syncthreads();

    v16h a[2], b[2];
#pragma unroll
    for (int sm = 0; sm < 2; ++sm) a[sm] = load_frag16(&As[0][0], wm + sm * 16 + l16, half);
#pragma unroll
    for (int sn = 0; sn < 2; ++sn) b[sn] = load_frag16(&Bs[0][0], wn + sn * 16 + l16, half);

#pragma unroll
    for (int sm = 0; sm < 2; ++sm)
#pragma unroll
      for (int sn = 0; sn < 2; ++sn)
        acc[sm][sn] = __builtin_amdgcn_wmma_f32_16x16x32_f16(
            false, a[sm], false, b[sn], (short)0, acc[sm][sn], false, false);
    __syncthreads();
  }

  // Store D + bias. C/D layout: VGPR r -> M = half*8 + r, N = l16.
#pragma unroll
  for (int sm = 0; sm < 2; ++sm)
#pragma unroll
    for (int sn = 0; sn < 2; ++sn) {
      int gn = tileN + wn + sn * 16 + l16;
      float bv = bias[gn];
#pragma unroll
      for (int r = 0; r < 8; ++r) {
        int gm = tileM + wm + sm * 16 + half * 8 + r;
        P[(size_t)gm * HID + gn] = acc[sm][sn][r] + bv;
      }
    }
}

// ---------------------------------------------------------------------------
// Phase 2: persistent recurrence. 64 WGs x 256 thr; each WG owns 32 rows of
// W_hh, held in LDS (32 x 2056 f32, padded -> conflict-free) for all steps.
// ---------------------------------------------------------------------------
#define NWG 64
#define TPB 256
#define RPW (HID / NWG)        // 32 rows per workgroup
#define LPR (TPB / RPW)        // 8 lanes per row
#define WSTR (HID + 8)         // padded LDS row stride (floats)
#define SMEM_FLOATS (RPW * WSTR + HID)
#define SMEM_BYTES (SMEM_FLOATS * 4)   // 271360 B < 320 KB

extern __shared__ float smem[];

__global__ __launch_bounds__(TPB)
void rnn_recur_kernel(const float* __restrict__ Whh,
                      float* __restrict__ out,   // SEQ x HID, pre-filled with P
                      float* __restrict__ hT,    // HID
                      float* __restrict__ hbuf,  // 2 x HID, zero-initialized
                      int*   __restrict__ ctr)   // zero-initialized
{
  float* Ws = smem;                 // RPW x WSTR
  float* hs = smem + RPW * WSTR;    // HID

  const int tid    = threadIdx.x;
  const int olocal = tid / LPR;                 // 0..31
  const int l8     = tid % LPR;                 // 0..7
  const int orow   = blockIdx.x * RPW + olocal; // global output row

  // Load this WG's 32 rows of W_hh into LDS once (64 float4 per thread).
  for (int i = tid; i < RPW * (HID / 4); i += TPB) {
    int r  = i / (HID / 4);
    int c4 = (i % (HID / 4)) * 4;
    float4 f = *(const float4*)(Whh + (size_t)(blockIdx.x * RPW + r) * HID + c4);
    float* d = Ws + r * WSTR + c4;
    d[0] = f.x; d[1] = f.y; d[2] = f.z; d[3] = f.w;
  }
  __syncthreads();

  int expected = 0;
  for (int t = 0; t < SEQ; ++t) {
    const float* hprev = hbuf + (t & 1) * HID;
    float*       hnext = hbuf + ((t + 1) & 1) * HID;

    // Stage h_{t-1} into LDS (8 KB broadcast).
    for (int i = tid; i < HID / 4; i += TPB) {
      float4 f = *(const float4*)(hprev + i * 4);
      hs[i * 4 + 0] = f.x; hs[i * 4 + 1] = f.y;
      hs[i * 4 + 2] = f.z; hs[i * 4 + 3] = f.w;
    }
    __syncthreads();

    const float* wrow = Ws + olocal * WSTR;
    float sum = 0.0f;
#pragma unroll 8
    for (int k = l8; k < HID; k += LPR)
      sum = fmaf(wrow[k], hs[k], sum);

    // Reduce across the 8 lanes of this row.
    sum += __shfl_xor(sum, 4, 8);
    sum += __shfl_xor(sum, 2, 8);
    sum += __shfl_xor(sum, 1, 8);

    if (l8 == 0) {
      size_t idx = (size_t)t * HID + orow;
      float v = tanhf(sum + out[idx]);   // out holds P[t] = Wxh x_t + b
      out[idx]    = v;
      hnext[orow] = v;
      if (t == SEQ - 1) hT[orow] = v;
    }

    // Grid-wide barrier (monotonic counter).
    __syncthreads();
    if (tid == 0) {
      __threadfence();
      atomicAdd(ctr, 1);
      expected += (int)gridDim.x;
      while (__hip_atomic_load(ctr, __ATOMIC_RELAXED, __HIP_MEMORY_SCOPE_AGENT) < expected)
        __builtin_amdgcn_s_sleep(2);
      __threadfence();
    }
    __syncthreads();
  }
}

// ---------------------------------------------------------------------------
__global__ void init_kernel(float* hbuf, int* ctr) {
  int i = blockIdx.x * blockDim.x + threadIdx.x;
  if (i < 2 * HID) hbuf[i] = 0.0f;
  if (i == 0) *ctr = 0;
}

extern "C" void kernel_launch(void* const* d_in, const int* in_sizes, int n_in,
                              void* d_out, int out_size, void* d_ws, size_t ws_size,
                              hipStream_t stream) {
  (void)in_sizes; (void)n_in; (void)out_size; (void)ws_size;
  const float* X   = (const float*)d_in[0];
  const float* Wxh = (const float*)d_in[1];
  const float* Whh = (const float*)d_in[2];
  const float* b   = (const float*)d_in[3];

  float* out  = (float*)d_out;                   // SEQ*HID outputs (tuple part 1)
  float* hT   = out + (size_t)SEQ * HID;         // final hidden (tuple part 2)
  float* hbuf = (float*)d_ws;                    // 2*HID floats
  int*   ctr  = (int*)((char*)d_ws + 2 * HID * sizeof(float));

  init_kernel<<<(2 * HID + 255) / 256, 256, 0, stream>>>(hbuf, ctr);

  dim3 g1(SEQ / BM, HID / BN);                   // 64 x 32 blocks
  gemm_xh_kernel<<<g1, 256, 0, stream>>>(X, Wxh, b, out);

  hipFuncSetAttribute(reinterpret_cast<const void*>(rnn_recur_kernel),
                      hipFuncAttributeMaxDynamicSharedMemorySize, SMEM_BYTES);
  rnn_recur_kernel<<<NWG, TPB, SMEM_BYTES, stream>>>(Whh, out, hT, hbuf, ctr);
}
